// MaskDecoder_27376121545520
// MI455X (gfx1250) — matrix operand order repaired and checked by reading, not compiled, not run on hardware
//
#include <hip/hip_runtime.h>

typedef __bf16 bf16_t;
typedef bf16_t v16bf __attribute__((ext_vector_type(16)));
typedef bf16_t v8bf  __attribute__((ext_vector_type(8)));
typedef float  v8f   __attribute__((ext_vector_type(8)));

#define KBANDS 31
#define CDIM   128
#define HDIM   512
#define TDIM   1000
#define E2DIM  54      // 2*M
#define E2PAD  64
#define MDIM   27
#define FDIM   257
#define TM     128     // rows per tile
#define BDIM   8

__constant__ int c_bands[KBANDS] = {3,3,3,3,3,3,3,3,3,3,
                                    6,6,6,6,6,6,6,6,6,6,6,6,
                                    16,16,16,16,16,16,16,16,
                                    27};
__constant__ int c_base[KBANDS]  = {0,3,6,9,12,15,18,21,24,27,
                                    30,36,42,48,54,60,66,72,78,84,90,96,
                                    102,118,134,150,166,182,198,214,
                                    230};

static __device__ __forceinline__ bf16_t f2bf(float f) { return (bf16_t)f; }

// Branchless tanh: tanh(x) = sign(x) * (1 - e^{-2|x|}) / (1 + e^{-2|x|}).
// Denominator in [1,2] -> raw v_rcp_f32 approximation is plenty for a bf16 result.
static __device__ __forceinline__ float tanh_fast(float x) {
  const float ax = __builtin_fabsf(x);
  const float t  = __expf(-2.f * ax);
  const float r  = (1.f - t) * __builtin_amdgcn_rcpf(1.f + t);
  return __builtin_copysignf(r, x);
}

static __device__ __forceinline__ float sigmoid_fast(float x) {
  return __builtin_amdgcn_rcpf(1.f + __expf(-x));
}

static __device__ __forceinline__ float wave_reduce_add(float v) {
  #pragma unroll
  for (int m = 16; m > 0; m >>= 1) v += __shfl_xor(v, m, 32);
  return v;
}

// A fragment: 16xK32 tile, row-major source with given stride (bf16 elems).
// lane L: row = m0 + (L&15); K chunks [k0+8*(L>>4), +8) and [k0+16+8*(L>>4), +8)
static __device__ __forceinline__ v16bf load_a_frag(const bf16_t* src, int stride,
                                                    int m0, int k0, int lane) {
  const int half = lane >> 4, r = lane & 15;
  const bf16_t* base = src + (size_t)(m0 + r) * stride + k0 + 8 * half;
  union { v16bf v; v8bf h[2]; } u;
  u.h[0] = *(const v8bf*)(base);
  u.h[1] = *(const v8bf*)(base + 16);
  return u.v;
}

// B fragment: K32x16 tile where B[k][n] = W[n][k], W row-major with given stride.
// lane L: col = n0 + (L&15); K chunks [k0+8*(L>>4), +8) and [k0+16+8*(L>>4), +8) in W's row.
static __device__ __forceinline__ v16bf load_b_frag(const bf16_t* w, int stride,
                                                    int n0, int k0, int lane) {
  const int half = lane >> 4, col = lane & 15;
  const bf16_t* base = w + (size_t)(n0 + col) * stride + k0 + 8 * half;
  union { v16bf v; v8bf h[2]; } u;
  u.h[0] = *(const v8bf*)(base);
  u.h[1] = *(const v8bf*)(base + 16);
  return u.v;
}

// ---------------- Kernel A: per-(b,k) mean / rsigma ----------------
__global__ __launch_bounds__(256)
void stats_kernel(const float* __restrict__ x, float* __restrict__ stats) {
  const int bk = blockIdx.x;                     // 0..247
  const float* xg = x + (size_t)bk * (TDIM * CDIM);
  float s = 0.f, ss = 0.f;
  for (int i = threadIdx.x; i < (TDIM * CDIM) / 4; i += 256) {
    float4 v = *(const float4*)(xg + (size_t)i * 4);
    s  += v.x + v.y + v.z + v.w;
    ss += v.x * v.x + v.y * v.y + v.z * v.z + v.w * v.w;
  }
  __shared__ float sh_s[8], sh_ss[8];
  s = wave_reduce_add(s);
  ss = wave_reduce_add(ss);
  if ((threadIdx.x & 31) == 0) { sh_s[threadIdx.x >> 5] = s; sh_ss[threadIdx.x >> 5] = ss; }
  __syncthreads();
  if (threadIdx.x == 0) {
    float ts = 0.f, tss = 0.f;
    #pragma unroll
    for (int i = 0; i < 8; ++i) { ts += sh_s[i]; tss += sh_ss[i]; }
    const float inv_n = 1.0f / (float)(TDIM * CDIM);
    float mu = ts * inv_n;
    float var = tss * inv_n - mu * mu;
    stats[2 * bk]     = mu;
    stats[2 * bk + 1] = rsqrtf(var + 1e-5f);
  }
}

// ---------------- Kernel B: weight-norm v1 -> bf16 W1 ----------------
__global__ __launch_bounds__(256)
void prep_w1(const float* __restrict__ v1, const float* __restrict__ g1,
             bf16_t* __restrict__ w1) {
  const int row  = blockIdx.x * 8 + (threadIdx.x >> 5);   // K*H rows
  const int lane = threadIdx.x & 31;
  const float* src = v1 + (size_t)row * CDIM;
  float4 v = *(const float4*)(src + lane * 4);
  float ss = v.x * v.x + v.y * v.y + v.z * v.z + v.w * v.w;
  ss = wave_reduce_add(ss);
  const float scale = g1[row] * rsqrtf(ss);
  bf16_t* d = w1 + (size_t)row * CDIM + lane * 4;
  d[0] = f2bf(v.x * scale); d[1] = f2bf(v.y * scale);
  d[2] = f2bf(v.z * scale); d[3] = f2bf(v.w * scale);
}

// ---------------- Kernel C: weight-norm v2 -> bf16 W2 (padded 54->64) ----------------
__global__ __launch_bounds__(256)
void prep_w2(const float* __restrict__ v2, const float* __restrict__ g2,
             bf16_t* __restrict__ w2) {
  const int row  = blockIdx.x * 8 + (threadIdx.x >> 5);   // K*64 rows
  const int lane = threadIdx.x & 31;
  const int k = row >> 6, e = row & 63;
  bf16_t* d = w2 + (size_t)row * HDIM + lane * 16;
  if (e < E2DIM) {
    const float* src = v2 + ((size_t)k * E2DIM + e) * HDIM + lane * 16;
    float vals[16];
    float ss = 0.f;
    #pragma unroll
    for (int i = 0; i < 4; ++i) {
      float4 v = *(const float4*)(src + i * 4);
      vals[4*i+0] = v.x; vals[4*i+1] = v.y; vals[4*i+2] = v.z; vals[4*i+3] = v.w;
      ss += v.x * v.x + v.y * v.y + v.z * v.z + v.w * v.w;
    }
    ss = wave_reduce_add(ss);
    const float scale = g2[k * E2DIM + e] * rsqrtf(ss);
    #pragma unroll
    for (int i = 0; i < 16; ++i) d[i] = f2bf(vals[i] * scale);
  } else {
    #pragma unroll
    for (int i = 0; i < 16; ++i) d[i] = f2bf(0.f);
  }
}

// ---------------- Main fused kernel ----------------
// LDS: [0, 128KB) h tile (128x512 bf16); [128KB, 160KB) union: xn (128x128 bf16) / o (128x64 f32)
#define LDS_H_BYTES  (TM * HDIM * 2)
#define LDS_TOTAL    (LDS_H_BYTES + TM * CDIM * 2)

__global__ __launch_bounds__(256)
void mask_decoder_main(const float* __restrict__ x,
                       const float* __restrict__ gnw, const float* __restrict__ gnb,
                       const float* __restrict__ b1,  const float* __restrict__ b2,
                       const float* __restrict__ stats,
                       const bf16_t* __restrict__ w1, const bf16_t* __restrict__ w2,
                       float* __restrict__ out) {
  extern __shared__ char smem[];
  bf16_t* lds_h  = (bf16_t*)smem;
  bf16_t* lds_xn = (bf16_t*)(smem + LDS_H_BYTES);
  float*  lds_o  = (float*)(smem + LDS_H_BYTES);   // aliases xn (disjoint lifetime)

  const int bk = blockIdx.y;
  const int k = bk % KBANDS, b = bk / KBANDS;
  const int t0 = blockIdx.x * TM;
  const int tid = threadIdx.x, w = tid >> 5, lane = tid & 31;
  const int half = lane >> 4, col = lane & 15;

  const float mu = stats[2 * bk], rsig = stats[2 * bk + 1];

  // ---- Stage 1: load + groupnorm x tile -> LDS bf16 ----
  const float* xg = x + (size_t)bk * (TDIM * CDIM);
  for (int i = tid; i < TM * (CDIM / 4); i += 256) {
    const int row = i >> 5;             // CDIM/4 = 32 float4 per row
    const int c = (i & 31) * 4;
    const int t = t0 + row;
    float4 v = make_float4(0.f, 0.f, 0.f, 0.f);
    if (t < TDIM) v = *(const float4*)(xg + (size_t)t * CDIM + c);
    const float4 gw = *(const float4*)(gnw + k * CDIM + c);
    const float4 gb = *(const float4*)(gnb + k * CDIM + c);
    bf16_t* d = lds_xn + row * CDIM + c;
    d[0] = f2bf((v.x - mu) * rsig * gw.x + gb.x);
    d[1] = f2bf((v.y - mu) * rsig * gw.y + gb.y);
    d[2] = f2bf((v.z - mu) * rsig * gw.z + gb.z);
    d[3] = f2bf((v.w - mu) * rsig * gw.w + gb.w);
  }
  __syncthreads();

  // ---- Stage 2: GEMM1 h = tanh(xn @ W1^T + b1)  [128x128 @ 128x512] ----
  const bf16_t* w1k = w1 + (size_t)k * HDIM * CDIM;
  for (int it = 0; it < 4; ++it) {
    const int n0 = (w + it * 8) * 16;          // wave partitions 32 N-tiles
    v16bf bfrag[4];
    #pragma unroll
    for (int kk = 0; kk < 4; ++kk) bfrag[kk] = load_b_frag(w1k, CDIM, n0, kk * 32, lane);
    const float b1v = b1[k * HDIM + n0 + col];
    for (int mt = 0; mt < 8; ++mt) {
      const int m0 = mt * 16;
      v8f acc;
      #pragma unroll
      for (int j = 0; j < 8; ++j) acc[j] = 0.f;
      #pragma unroll
      for (int kk = 0; kk < 4; ++kk) {
        v16bf a = load_a_frag(lds_xn, CDIM, m0, kk * 32, lane);
        acc = __builtin_amdgcn_wmma_f32_16x16x32_bf16(false, a, false, bfrag[kk],
                                                      (short)0, acc, false, false);
      }
      #pragma unroll
      for (int j = 0; j < 8; ++j) {
        const int r = m0 + j + 8 * half;
        lds_h[r * HDIM + n0 + col] = f2bf(tanh_fast(acc[j] + b1v));
      }
    }
  }
  __syncthreads();

  // ---- Stage 3: GEMM2 o = h @ W2^T + b2   [128x512 @ 512x64] ----
  const bf16_t* w2k = w2 + (size_t)k * E2PAD * HDIM;
  const int n0 = (w & 3) * 16;                 // 4 N-tiles over 8 waves
  const int mbase = (w >> 2);                  // m-tiles {mbase, mbase+2, +4, +6}
  const int e = n0 + col;
  const float b2v = (e < E2DIM) ? b2[k * E2DIM + e] : 0.f;
  v8f acc2[4];
  #pragma unroll
  for (int i = 0; i < 4; ++i)
    #pragma unroll
    for (int j = 0; j < 8; ++j) acc2[i][j] = 0.f;
  v16bf bnext = load_b_frag(w2k, HDIM, n0, 0, lane);   // software pipeline B loads
  #pragma unroll
  for (int kk = 0; kk < 16; ++kk) {
    v16bf bfrag = bnext;
    if (kk < 15) bnext = load_b_frag(w2k, HDIM, n0, (kk + 1) * 32, lane);
    #pragma unroll
    for (int i = 0; i < 4; ++i) {
      const int m0 = (mbase + 2 * i) * 16;
      v16bf a = load_a_frag(lds_h, HDIM, m0, kk * 32, lane);
      acc2[i] = __builtin_amdgcn_wmma_f32_16x16x32_bf16(false, a, false, bfrag,
                                                        (short)0, acc2[i], false, false);
    }
  }
  #pragma unroll
  for (int i = 0; i < 4; ++i) {
    const int m0 = (mbase + 2 * i) * 16;
    #pragma unroll
    for (int j = 0; j < 8; ++j)
      lds_o[(m0 + j + 8 * half) * E2PAD + e] = acc2[i][j] + b2v;
  }
  __syncthreads();

  // ---- Stage 4: gated sigmoid + banded gather ----
  const int nb = c_bands[k], fb = c_base[k];
  const int valid_t = min(TM, TDIM - t0);
  float* og = out + ((size_t)b * FDIM + fb) * TDIM + t0;
  for (int i = tid; i < valid_t * nb; i += 256) {
    const int t = i / nb, e2 = i - t * nb;
    const float ao = lds_o[t * E2PAD + e2];
    const float go = lds_o[t * E2PAD + MDIM + e2];
    og[(size_t)e2 * TDIM + t] = 2.f * sigmoid_fast(ao * sigmoid_fast(go));
  }
}

extern "C" void kernel_launch(void* const* d_in, const int* in_sizes, int n_in,
                              void* d_out, int out_size, void* d_ws, size_t ws_size,
                              hipStream_t stream) {
  (void)in_sizes; (void)n_in; (void)out_size; (void)ws_size;
  const float* x    = (const float*)d_in[0];
  const float* gn_w = (const float*)d_in[1];
  const float* gn_b = (const float*)d_in[2];
  const float* v1   = (const float*)d_in[3];
  const float* g1   = (const float*)d_in[4];
  const float* b1   = (const float*)d_in[5];
  const float* v2   = (const float*)d_in[6];
  const float* g2   = (const float*)d_in[7];
  const float* b2   = (const float*)d_in[8];
  float* out = (float*)d_out;

  char* ws = (char*)d_ws;
  float*  stats = (float*)ws;                                   // 248*2 floats
  bf16_t* w1bf  = (bf16_t*)(ws + 4096);                         // 31*512*128 bf16
  bf16_t* w2bf  = (bf16_t*)(ws + 4096 + (size_t)KBANDS * HDIM * CDIM * 2); // 31*64*512 bf16

  stats_kernel<<<dim3(BDIM * KBANDS), 256, 0, stream>>>(x, stats);
  prep_w1<<<dim3(KBANDS * HDIM / 8), 256, 0, stream>>>(v1, g1, w1bf);
  prep_w2<<<dim3(KBANDS * E2PAD / 8), 256, 0, stream>>>(v2, g2, w2bf);

  (void)hipFuncSetAttribute((const void*)mask_decoder_main,
                            hipFuncAttributeMaxDynamicSharedMemorySize, LDS_TOTAL);
  mask_decoder_main<<<dim3((TDIM + TM - 1) / TM, BDIM * KBANDS), 256, LDS_TOTAL, stream>>>(
      x, gn_w, gn_b, b1, b2, stats, w1bf, w2bf, out);
}
